// DeformableCrossAttention_20332375179633
// MI455X (gfx1250) — matrix operand order repaired and checked by reading, not compiled, not run on hardware
//
#include <hip/hip_runtime.h>
#include <hip/hip_bf16.h>
#include <math.h>

typedef __attribute__((ext_vector_type(16))) _Float16 v16h;
typedef __attribute__((ext_vector_type(8)))  _Float16 v8h;
typedef __attribute__((ext_vector_type(4)))  _Float16 v4h;
typedef __attribute__((ext_vector_type(8)))  float    v8f;

// Fixed problem geometry (all powers of two -> div/mod fold to shifts/masks)
#define BN   2
#define HH   256
#define WW   256
#define DIMC 128
#define KPTS 4
#define NPIX ((long)BN * HH * WW)      // 131072
#define PXS  (DIMC + 8)   // LDS pixel stride in halves: 272B breaks bank aliasing

__device__ __forceinline__ float gelu_erf(float v) {
    return 0.5f * v * (1.0f + erff(v * 0.70710678118654752f));
}

// ---------------------------------------------------------------------------
// Pack conv weights: fp32 OIHW -> f16 [CPAD][(ky*KW+kx)*CIN + ci], zero pad
// rows COUT..CPAD-1. This K ordering makes WMMA A fragments contiguous.
// ---------------------------------------------------------------------------
template <int COUT, int CIN, int KK, int CPAD>
__global__ void pack_weights_kernel(const float* __restrict__ src,
                                    _Float16* __restrict__ dst) {
    const int Ktot = CIN * KK * KK;
    long i = (long)blockIdx.x * blockDim.x + threadIdx.x;
    if (i >= (long)CPAD * Ktot) return;
    int k  = (int)(i % Ktot);
    int co = (int)(i / Ktot);
    int ci = k % CIN;
    int t  = k / CIN;
    int kx = t % KK;
    int ky = t / KK;
    float v = 0.0f;
    if (co < COUT)
        v = src[(((long)co * CIN + ci) * KK + ky) * KK + kx];
    dst[i] = (_Float16)v;
}

// ---------------------------------------------------------------------------
// NCHW fp32 -> NHWC f16 (fixed geometry: shifts/masks only)
// ---------------------------------------------------------------------------
__global__ void to_nhwc_half_kernel(const float* __restrict__ src,
                                    _Float16* __restrict__ dst) {
    long i = (long)blockIdx.x * blockDim.x + threadIdx.x;   // grid sized exactly
    int c = (int)(i & (DIMC - 1));
    int w = (int)((i >> 7) & (WW - 1));
    int h = (int)((i >> 15) & (HH - 1));
    int b = (int)(i >> 23);
    dst[i] = (_Float16)src[(((long)b * DIMC + c) * HH + h) * WW + w];
}

// ---------------------------------------------------------------------------
// Implicit-GEMM conv (Cin = 128) via V_WMMA_F32_16X16X32_F16.
// Block = GROUPS x MT waves. Each group owns one 16-pixel tile; its MT waves
// cooperatively stage the halo'd input tile into LDS (zero pad resolved here,
// once), then run a branch-free ds_load + wmma loop.
//   A frag (16-bit A 16x32): lane<16 -> M=lane, K0-15; lane>=16 ->
//     M=lane-16, K16-31 => one contiguous 32B run (two 16B loads) of the weight row.
//   B frag: lane<16 col N=lane K0-15; lane>=16 K16-31 => one contiguous 32B
//     run of staged channels.
//   C/D: VGPR r -> M = r (+8 for lane>=16), N = lane&15.
// ---------------------------------------------------------------------------
template <int K3, int MT, int GROUPS, int COUTSTORE, int GELU>
__global__ void wmma_conv_kernel(const _Float16* __restrict__ in,   // [BN,H,W,128]
                                 const _Float16* __restrict__ wp,   // [16*MT][Ktot]
                                 const float* __restrict__ bias,    // [COUTSTORE]
                                 _Float16* __restrict__ out) {      // [BN,H,W,COUTSTORE]
    constexpr int ROWS = K3 ? 3 : 1;
    constexpr int COLS = K3 ? 18 : 16;
    constexpr int AREA = ROWS * COLS * PXS;             // halves per group
    constexpr int Ktot = DIMC * ROWS * ROWS;            // 1152 or 128
    __shared__ _Float16 ldsbuf[GROUPS * AREA];

    const int lane  = threadIdx.x & 31;
    const int col   = lane & 15;
    const int hi    = lane >> 4;
    const int wave  = threadIdx.x >> 5;
    const int group = wave / MT;
    const int mT    = wave % MT;

    long ptile = (long)blockIdx.x * GROUPS + group;     // grid sized exactly
    const int xT = (int)(ptile & 15);                   // W/16 = 16 tiles
    const int y  = (int)((ptile >> 4) & (HH - 1));
    const int b  = (int)(ptile >> 12);
    const int x0 = xT << 4;
    const int m0 = mT << 4;

    _Float16* area = ldsbuf + group * AREA;

    // ---- cooperative stage: [ROWS][COLS][128] (+pad) with zero halo -------
    {
        const int tIdx = threadIdx.x - group * MT * 32;
        constexpr int NTHR = MT * 32;
        constexpr int NVEC = ROWS * COLS * (DIMC / 8);  // v8h elements
#pragma unroll
        for (int i = tIdx; i < NVEC; i += NTHR) {
            int cv = i & 15;                            // 128/8 = 16
            int p  = (i >> 4) % COLS;
            int r  = (i >> 4) / COLS;
            int gy = K3 ? (y + r - 1) : y;
            int gx = K3 ? (x0 + p - 1) : (x0 + p);
            v8h v = {};
            if (gy >= 0 && gy < HH && gx >= 0 && gx < WW) {
                const _Float16* sp = in + (((long)b * HH + gy) * WW + gx) * DIMC + cv * 8;
                __builtin_prefetch(sp + (long)WW * DIMC, 0, 0);  // next-row hint
                v = *(const v8h*)sp;
            }
            *(v8h*)(area + (r * COLS + p) * PXS + cv * 8) = v;
        }
    }
    __syncthreads();

    // ---- branch-free WMMA loop --------------------------------------------
    const _Float16* arow = wp + (long)(m0 + col) * Ktot + (hi ? 16 : 0);

    v8f acc = {};
#pragma unroll
    for (int ky = 0; ky < ROWS; ++ky) {
#pragma unroll
        for (int kx = 0; kx < ROWS; ++kx) {
            const _Float16* lb = area + (ky * COLS + col + kx) * PXS + hi * 16;
            const _Float16* ab = arow + (ky * ROWS + kx) * DIMC;
#pragma unroll
            for (int c0 = 0; c0 < DIMC; c0 += 32) {
                v16h bfrag = *(const v16h*)(lb + c0);   // 2x ds_load_b128
                v8h alo = *(const v8h*)(ab + c0);
                v8h ahi = *(const v8h*)(ab + c0 + 8);
                v16h afrag;
#pragma unroll
                for (int i = 0; i < 8; ++i) { afrag[i] = alo[i]; afrag[i + 8] = ahi[i]; }
                acc = __builtin_amdgcn_wmma_f32_16x16x32_f16(
                    false, afrag, false, bfrag, (short)0, acc, false, false);
            }
        }
    }

    long pix = ((long)b * HH + y) * WW + (x0 + col);
#pragma unroll
    for (int r = 0; r < 8; ++r) {
        int m = m0 + r + (hi ? 8 : 0);
        if (m < COUTSTORE) {                            // compile-time folds
            float v = acc[r] + bias[m];
            if (GELU) v = gelu_erf(v);
            out[pix * COUTSTORE + m] = (_Float16)v;
        }
    }
}

// ---------------------------------------------------------------------------
// weight_net tail: 1x1 conv 32->4 + softmax over K, per pixel.
// ---------------------------------------------------------------------------
__global__ void logits_softmax_kernel(const _Float16* __restrict__ whid, // [pix][32]
                                      const float* __restrict__ w2,      // [4][32]
                                      const float* __restrict__ b2,      // [4]
                                      float* __restrict__ wsoft) {       // [pix][4]
    long p = (long)blockIdx.x * blockDim.x + threadIdx.x;   // grid exact
    const _Float16* f = whid + p * 32;
    float lg[4];
#pragma unroll
    for (int k = 0; k < 4; ++k) {
        float s = b2[k];
#pragma unroll
        for (int c = 0; c < 32; ++c) s += (float)f[c] * w2[k * 32 + c];
        lg[k] = s;
    }
    float mx = fmaxf(fmaxf(lg[0], lg[1]), fmaxf(lg[2], lg[3]));
    float e0 = expf(lg[0] - mx), e1 = expf(lg[1] - mx);
    float e2 = expf(lg[2] - mx), e3 = expf(lg[3] - mx);
    float inv = 1.0f / (e0 + e1 + e2 + e3);
    wsoft[p * 4 + 0] = e0 * inv;
    wsoft[p * 4 + 1] = e1 * inv;
    wsoft[p * 4 + 2] = e2 * inv;
    wsoft[p * 4 + 3] = e3 * inv;
}

// ---------------------------------------------------------------------------
// Deformable sampling + weighted aggregation. Wave per pixel; lane owns 4 ch.
// Border-clamped bilinear on NHWC f16 key (contiguous b64 per corner fetch).
// ---------------------------------------------------------------------------
__global__ void sample_agg_kernel(const _Float16* __restrict__ key,  // [B,H,W,128]
                                  const _Float16* __restrict__ offs, // [pix][8] (x,y)*K
                                  const float* __restrict__ wsoft,   // [pix][4]
                                  _Float16* __restrict__ agg) {      // [pix][128]
    const int lane = threadIdx.x & 31;
    long pix = (long)blockIdx.x * (blockDim.x >> 5) + (threadIdx.x >> 5); // grid exact
    const int x = (int)(pix & (WW - 1));
    const int y = (int)((pix >> 8) & (HH - 1));
    const int b = (int)(pix >> 16);
    const int ch = lane * 4;
    const _Float16* kb = key + (long)b * HH * WW * DIMC;

    float a0 = 0.f, a1 = 0.f, a2 = 0.f, a3 = 0.f;
    float gxb = -1.0f + 2.0f * (float)x * (1.0f / (WW - 1));
    float gyb = -1.0f + 2.0f * (float)y * (1.0f / (HH - 1));
#pragma unroll
    for (int k = 0; k < KPTS; ++k) {
        float ox = (float)offs[pix * 8 + 2 * k];
        float oy = (float)offs[pix * 8 + 2 * k + 1];
        float gx = fminf(fmaxf(gxb + ox * (2.0f / WW), -1.0f), 1.0f);
        float gy = fminf(fmaxf(gyb + oy * (2.0f / HH), -1.0f), 1.0f);
        float px = fminf(fmaxf((gx + 1.0f) * 0.5f * (WW - 1), 0.0f), (float)(WW - 1));
        float py = fminf(fmaxf((gy + 1.0f) * 0.5f * (HH - 1), 0.0f), (float)(HH - 1));
        int x0 = (int)px;              // px >= 0 -> trunc == floor
        int y0 = (int)py;
        int x1 = (x0 + 1 < WW) ? x0 + 1 : WW - 1;
        int y1 = (y0 + 1 < HH) ? y0 + 1 : HH - 1;
        float wx = px - (float)x0, wy = py - (float)y0;
        float wk = wsoft[pix * 4 + k];
        float w00 = (1.f - wx) * (1.f - wy) * wk;
        float w01 = wx * (1.f - wy) * wk;
        float w10 = (1.f - wx) * wy * wk;
        float w11 = wx * wy * wk;
        v4h p00 = *(const v4h*)(kb + ((long)y0 * WW + x0) * DIMC + ch);
        v4h p01 = *(const v4h*)(kb + ((long)y0 * WW + x1) * DIMC + ch);
        v4h p10 = *(const v4h*)(kb + ((long)y1 * WW + x0) * DIMC + ch);
        v4h p11 = *(const v4h*)(kb + ((long)y1 * WW + x1) * DIMC + ch);
        a0 += w00 * (float)p00[0] + w01 * (float)p01[0] + w10 * (float)p10[0] + w11 * (float)p11[0];
        a1 += w00 * (float)p00[1] + w01 * (float)p01[1] + w10 * (float)p10[1] + w11 * (float)p11[1];
        a2 += w00 * (float)p00[2] + w01 * (float)p01[2] + w10 * (float)p10[2] + w11 * (float)p11[2];
        a3 += w00 * (float)p00[3] + w01 * (float)p01[3] + w10 * (float)p10[3] + w11 * (float)p11[3];
    }
    v4h r;
    r[0] = (_Float16)a0; r[1] = (_Float16)a1; r[2] = (_Float16)a2; r[3] = (_Float16)a3;
    *(v4h*)(agg + pix * DIMC + ch) = r;
}

// ---------------------------------------------------------------------------
// Final 1x1 GEMM (fw2) + residual: out_nchw = query + 0.3 * (W2 * fhid + b2)
// Block: 8 waves = 8 cout tiles of one 16-pixel tile; B staged in LDS once.
// ---------------------------------------------------------------------------
__global__ void wmma_fuse2_kernel(const _Float16* __restrict__ in,  // [pix][128]
                                  const _Float16* __restrict__ wp,  // [128][128]
                                  const float* __restrict__ bias,
                                  const float* __restrict__ query,  // NCHW f32
                                  float* __restrict__ out) {        // NCHW f32
    __shared__ _Float16 s[16 * PXS];
    const int lane = threadIdx.x & 31;
    const int col  = lane & 15;
    const int hi   = lane >> 4;
    const int mT   = threadIdx.x >> 5;          // 8 waves = 8 cout tiles
    long p0 = (long)blockIdx.x << 4;

    {   // stage 16 pixels x 128 ch (one v8h per thread)
        int i  = threadIdx.x;
        int cv = i & 15;
        int p  = i >> 4;
        v8h v = *(const v8h*)(in + (p0 + p) * DIMC + cv * 8);
        *(v8h*)(s + p * PXS + cv * 8) = v;
    }
    __syncthreads();

    int m0 = mT << 4;
    const _Float16* arow = wp + (long)(m0 + col) * DIMC + (hi ? 16 : 0);
    const _Float16* lb   = s + col * PXS + hi * 16;

    v8f acc = {};
#pragma unroll
    for (int c0 = 0; c0 < DIMC; c0 += 32) {
        v16h bfrag = *(const v16h*)(lb + c0);
        v8h alo = *(const v8h*)(arow + c0);
        v8h ahi = *(const v8h*)(arow + c0 + 8);
        v16h afrag;
#pragma unroll
        for (int i = 0; i < 8; ++i) { afrag[i] = alo[i]; afrag[i + 8] = ahi[i]; }
        acc = __builtin_amdgcn_wmma_f32_16x16x32_f16(
            false, afrag, false, bfrag, (short)0, acc, false, false);
    }

    long pix  = p0 + col;
    long bimg = pix >> 16;                      // H*W = 65536
    long hw   = pix & 65535;
#pragma unroll
    for (int r = 0; r < 8; ++r) {
        int m = m0 + r + (hi ? 8 : 0);
        long idx = ((bimg << 7) + m) * (HH * WW) + hw;   // (bimg*128 + m)*HW
        out[idx] = query[idx] + 0.3f * (acc[r] + bias[m]);
    }
}

// ---------------------------------------------------------------------------
extern "C" void kernel_launch(void* const* d_in, const int* in_sizes, int n_in,
                              void* d_out, int out_size, void* d_ws, size_t ws_size,
                              hipStream_t stream) {
    (void)in_sizes; (void)n_in; (void)out_size; (void)ws_size;
    const float* query = (const float*)d_in[0];
    const float* keyf  = (const float*)d_in[1];
    const float* ow1 = (const float*)d_in[2];
    const float* ob1 = (const float*)d_in[3];
    const float* ow2 = (const float*)d_in[4];
    const float* ob2 = (const float*)d_in[5];
    const float* ww1 = (const float*)d_in[6];
    const float* wb1 = (const float*)d_in[7];
    const float* ww2 = (const float*)d_in[8];
    const float* wb2 = (const float*)d_in[9];
    const float* fw1 = (const float*)d_in[10];
    const float* fb1 = (const float*)d_in[11];
    const float* fw2 = (const float*)d_in[12];
    const float* fb2 = (const float*)d_in[13];
    float* outp = (float*)d_out;

    char* ws = (char*)d_ws;
    size_t off = 0;
    auto alloc = [&](size_t bytes) -> char* {
        char* p = ws + off;
        off = (off + bytes + 255) & ~(size_t)255;
        return p;
    };
    _Float16* qh   = (_Float16*)alloc((size_t)NPIX * DIMC * 2);   // reused as agg
    _Float16* kh   = (_Float16*)alloc((size_t)NPIX * DIMC * 2);
    _Float16* hid1 = (_Float16*)alloc((size_t)NPIX * DIMC * 2);   // reused as fhid
    _Float16* whid = (_Float16*)alloc((size_t)NPIX * 32 * 2);
    _Float16* offsb= (_Float16*)alloc((size_t)NPIX * 8 * 2);
    float*    wsof = (float*)   alloc((size_t)NPIX * 4 * 4);
    _Float16* ow1p = (_Float16*)alloc((size_t)128 * 1152 * 2);
    _Float16* ow2p = (_Float16*)alloc((size_t)16  * 1152 * 2);
    _Float16* ww1p = (_Float16*)alloc((size_t)32  * 1152 * 2);
    _Float16* fw1p = (_Float16*)alloc((size_t)128 * 128 * 2);
    _Float16* fw2p = (_Float16*)alloc((size_t)128 * 128 * 2);
    _Float16* agg  = qh;
    _Float16* fhid = hid1;

    // ---- weight packing ----------------------------------------------------
    pack_weights_kernel<128, 128, 3, 128><<<(128 * 1152 + 255) / 256, 256, 0, stream>>>(ow1, ow1p);
    pack_weights_kernel<8,   128, 3, 16 ><<<(16 * 1152 + 255) / 256,  256, 0, stream>>>(ow2, ow2p);
    pack_weights_kernel<32,  128, 3, 32 ><<<(32 * 1152 + 255) / 256,  256, 0, stream>>>(ww1, ww1p);
    pack_weights_kernel<128, 128, 1, 128><<<(128 * 128 + 255) / 256,  256, 0, stream>>>(fw1, fw1p);
    pack_weights_kernel<128, 128, 1, 128><<<(128 * 128 + 255) / 256,  256, 0, stream>>>(fw2, fw2p);

    // ---- layout transform --------------------------------------------------
    {
        int blk = (int)(NPIX * DIMC / 256);     // exact
        to_nhwc_half_kernel<<<blk, 256, 0, stream>>>(query, qh);
        to_nhwc_half_kernel<<<blk, 256, 0, stream>>>(keyf, kh);
    }

    // ---- convs (pixel tiles = 8192, divisible by all GROUPS) ---------------
    wmma_conv_kernel<1, 8, 1, 128, 1><<<8192, 256, 0, stream>>>(qh,   ow1p, ob1, hid1);
    wmma_conv_kernel<1, 1, 4, 8,   0><<<2048, 128, 0, stream>>>(hid1, ow2p, ob2, offsb);
    wmma_conv_kernel<1, 2, 4, 32,  1><<<2048, 256, 0, stream>>>(qh,   ww1p, wb1, whid);

    logits_softmax_kernel<<<(int)(NPIX / 256), 256, 0, stream>>>(whid, ww2, wb2, wsof);
    sample_agg_kernel<<<(int)(NPIX / 8), 256, 0, stream>>>(kh, offsb, wsof, agg);

    wmma_conv_kernel<0, 8, 1, 128, 1><<<8192, 256, 0, stream>>>(agg, fw1p, fb1, fhid);

    wmma_fuse2_kernel<<<(int)(NPIX / 16), 256, 0, stream>>>(fhid, fw2p, fb2, query, outp);
}
